// EgoNet_59450937311402
// MI455X (gfx1250) — matrix operand order repaired and checked by reading, not compiled, not run on hardware
//
#include <hip/hip_runtime.h>
#include <math.h>

// ---------------------------------------------------------------------------
// EgoNet cost-volume + MLP for MI455X (gfx1250, wave32, WMMA).
//
// heat[br, e] = (1/(32+1e-6)) * sum over valid (c,d,h,w) of
//               feat0[br,c,d,h,w] * feat1[br,c,d+dz,h+dy,w+dx]
// mapped to banded 32x32 GEMMs (K = c,h,w) on v_wmma_f32_16x16x32_bf16 with
// hi/lo bf16 split (3 products) for near-fp32 accuracy.
// ---------------------------------------------------------------------------

typedef __bf16  bf16_t;
typedef bf16_t  v16bf __attribute__((ext_vector_type(16)));
typedef float   v8f   __attribute__((ext_vector_type(8)));

#define BRN 16
#define CN  32
#define DN  32
#define HN  32
#define WN  32
#define EN  243          // 9*3*9
#define RN  8
#define BN  2            // BRN / RN
#define NCOMBO 27        // (dy,dx) pairs: 3*9
#define HID 128
#define FIN (RN*EN)      // 1944
#define EPSF 1e-6f

// ---------------------------------------------------------------------------
// Kernel 1: correlation heat map via WMMA.
// grid = BRN*NCOMBO workgroups, 256 threads (8 wave32s).
// Each WG: fixed (br, dy, dx). Computes g[d][d'] = sum_k A*B over K=(c,h,w),
// then the 9 dz band sums -> heat[br][e] (disjoint per WG: plain stores).
// Row pointers advance by workgroup-uniform deltas (no per-chunk address
// rebuild); B-side dx mask/clamp is hoisted out of the loop.
// ---------------------------------------------------------------------------
__global__ __launch_bounds__(256) void corr_heat_kernel(
    const float* __restrict__ f0, const float* __restrict__ f1,
    float* __restrict__ heat) {

  const int bid   = blockIdx.x;
  const int br    = bid / NCOMBO;
  const int combo = bid % NCOMBO;
  const int yi = combo / 9, xi = combo % 9;
  const int dy = yi - 1,    dx = xi - 4;

  const int h0 = (dy < 0) ? -dy : 0;
  const int h1 = (dy > 0) ? (HN - dy) : HN;
  const int Hv = h1 - h0;                 // 31 or 32

  const int tid  = threadIdx.x;
  const int lane = tid & 31;
  const int wave = tid >> 5;
  const int m    = lane & 15;             // M (A-row / B-col) within tile
  const int sel  = lane >> 4;             // lane half selects K sub-range

  __shared__ float gws[8][DN * DN];       // per-wave partial g, 32 KB

  v8f acc[2][2];
  #pragma unroll
  for (int t = 0; t < 2; ++t)
    #pragma unroll
    for (int u = 0; u < 2; ++u)
      #pragma unroll
      for (int v = 0; v < 8; ++v) acc[t][u][v] = 0.f;

  const int base = br * CN * DN * HN * WN;        // dwords (< 2^31)

  // ---- loop-invariant per-lane state --------------------------------------
  // A element byte offsets are the compile-time constants i*4
  // once sel*16 dwords are folded into the row offset.
  int voffA[2], voffB[2];                          // byte offsets
  #pragma unroll
  for (int t = 0; t < 2; ++t)
    voffA[t] = (base + ((t * 16 + m) * HN + (h0 + wave)) * WN + sel * 16) * 4;
  #pragma unroll
  for (int u = 0; u < 2; ++u)
    voffB[u] = (base + ((u * 16 + m) * HN + (h0 + wave + dy)) * WN
                + sel * 16 + dx) * 4;

  // B-side dx validity mask + clamped element offsets (per lane, invariant).
  // Clamped address stays inside the current row -> always in-bounds.
  int   cOffB[16];
  float mkf[16];
  #pragma unroll
  for (int i = 0; i < 16; ++i) {
    const int wp = (sel << 4) + i + dx;
    const int wc = wp < 0 ? 0 : (wp > WN - 1 ? WN - 1 : wp);
    cOffB[i] = (wc - ((sel << 4) + dx)) * 4;       // bytes rel. to voffB
    mkf[i]   = (wp >= 0 && wp < WN) ? 1.f : 0.f;
  }

  const int deltaStep = 8 * WN * 4;                         // h += 8
  const int deltaWrap = (DN * HN + 8 - Hv) * WN * 4;        // c += 1, h wraps

  int hh = wave;                                   // wave < 8 <= Hv always
  for (int ch = wave; ch < CN * Hv; ch += 8) {
    v16bf ahi[2], alo[2], bhi[2], blo[2];

    // A (feat0): 16 loads at immediate offsets (4x b128 per tile)
    #pragma unroll
    for (int t = 0; t < 2; ++t) {
      const float* pA = (const float*)((const char*)f0 + voffA[t]);
      __builtin_prefetch((const char*)pA + deltaStep, 0, 1);
      #pragma unroll
      for (int i = 0; i < 16; ++i) {
        const float v = pA[i];
        const bf16_t hi = (bf16_t)v;
        ahi[t][i] = hi;
        alo[t][i] = (bf16_t)(v - (float)hi);
      }
    }
    // B (feat1): element i -> K = sel*16+i, w' = K+dx masked to [0,WN)
    #pragma unroll
    for (int u = 0; u < 2; ++u) {
      const char* pB = (const char*)f1 + voffB[u];
      __builtin_prefetch(pB + deltaStep, 0, 1);
      #pragma unroll
      for (int i = 0; i < 16; ++i) {
        const float v = *(const float*)(pB + cOffB[i]) * mkf[i];
        const bf16_t hi = (bf16_t)v;
        bhi[u][i] = hi;
        blo[u][i] = (bf16_t)(v - (float)hi);
      }
    }

    // 2x2 tiles, hi/lo split: hi*hi + hi*lo + lo*hi  (12 WMMAs / chunk)
    #pragma unroll
    for (int t = 0; t < 2; ++t)
      #pragma unroll
      for (int u = 0; u < 2; ++u) {
        acc[t][u] = __builtin_amdgcn_wmma_f32_16x16x32_bf16(
            false, ahi[t], false, bhi[u], (short)0, acc[t][u], false, false);
        acc[t][u] = __builtin_amdgcn_wmma_f32_16x16x32_bf16(
            false, ahi[t], false, blo[u], (short)0, acc[t][u], false, false);
        acc[t][u] = __builtin_amdgcn_wmma_f32_16x16x32_bf16(
            false, alo[t], false, bhi[u], (short)0, acc[t][u], false, false);
      }

    // advance (c,h) by 8 rows with wrap; uniform scalar delta
    hh += 8;
    int delta = deltaStep;
    if (hh >= Hv) { hh -= Hv; delta = deltaWrap; }  // stride 8 < Hv: <=1 wrap
    voffA[0] += delta; voffA[1] += delta;
    voffB[0] += delta; voffB[1] += delta;
  }

  // D layout: (vgpr v, lane) -> row M = v + 8*sel (+16t), col N = m (+16u)
  #pragma unroll
  for (int t = 0; t < 2; ++t)
    #pragma unroll
    for (int u = 0; u < 2; ++u)
      #pragma unroll
      for (int v = 0; v < 8; ++v)
        gws[wave][(t * 16 + v + sel * 8) * DN + (u * 16 + m)] = acc[t][u][v];

  __syncthreads();

  // band sums: heat[dz] = sum_d g[d][d+dz]; fixed-order adds (deterministic)
  if (tid < 9) {
    const int dz = tid - 4;
    float s = 0.f;
    for (int d = 0; d < DN; ++d) {
      const int j = d + dz;
      if (j >= 0 && j < DN) {
        float e = 0.f;
        #pragma unroll
        for (int wv = 0; wv < 8; ++wv) e += gws[wv][d * DN + j];
        s += e;
      }
    }
    const float scale = 1.0f / (EPSF + (float)CN);
    heat[br * EN + yi * 81 + xi * 9 + (dz + 4)] = s * scale;
  }
}

// ---------------------------------------------------------------------------
// Kernel 2: leaky-relu, L2 normalize, 3-layer MLP. Tiny (~1 MFLOP): 1 block.
// ---------------------------------------------------------------------------
__global__ __launch_bounds__(256) void mlp_kernel(
    const float* __restrict__ heat,
    const float* __restrict__ W1, const float* __restrict__ b1,
    const float* __restrict__ W2, const float* __restrict__ b2,
    const float* __restrict__ W3, const float* __restrict__ b3,
    float* __restrict__ out) {

  __shared__ float feat[BN * FIN];     // 3888 floats
  __shared__ float red0[256], red1[256];
  __shared__ float inv[BN];
  __shared__ float h1s[BN * HID], h2s[BN * HID];

  const int tid = threadIdx.x;

  // heat[BR][E] flat == feat[B][R*E] flat (same linearization)
  for (int idx = tid; idx < BN * FIN; idx += 256) {
    float x = heat[idx];
    feat[idx] = (x >= 0.f) ? x : 0.1f * x;
  }
  __syncthreads();

  float p0 = 0.f, p1 = 0.f;
  for (int idx = tid; idx < FIN; idx += 256) { float x = feat[idx];        p0 += x * x; }
  for (int idx = tid; idx < FIN; idx += 256) { float x = feat[FIN + idx];  p1 += x * x; }
  red0[tid] = p0; red1[tid] = p1;
  __syncthreads();
  if (tid == 0) {
    float s0 = 0.f, s1 = 0.f;
    for (int i = 0; i < 256; ++i) { s0 += red0[i]; s1 += red1[i]; }  // fixed order
    inv[0] = 1.0f / (EPSF + sqrtf(s0));
    inv[1] = 1.0f / (EPSF + sqrtf(s1));
  }
  __syncthreads();

  // layer 1: [B,1944] x W1^T -> [B,128] (normalization folded in post-dot)
  {
    const int o = tid & (HID - 1);
    const int b = tid >> 7;
    const float* wrow = W1 + (size_t)o * FIN;
    const float* fr = feat + b * FIN;
    float d = 0.f;
    for (int j = 0; j < FIN; ++j) d += fr[j] * wrow[j];
    float x = d * inv[b] + b1[o];
    h1s[b * HID + o] = (x >= 0.f) ? x : 0.1f * x;
  }
  __syncthreads();

  // layer 2
  {
    const int o = tid & (HID - 1);
    const int b = tid >> 7;
    const float* wrow = W2 + (size_t)o * HID;
    const float* fr = h1s + b * HID;
    float d = 0.f;
    #pragma unroll 4
    for (int j = 0; j < HID; ++j) d += fr[j] * wrow[j];
    float x = d + b2[o];
    h2s[b * HID + o] = (x >= 0.f) ? x : 0.1f * x;
  }
  __syncthreads();

  // layer 3: [B,128] x W3^T -> [B,4]
  if (tid < BN * 4) {
    const int b = tid >> 2;
    const int k = tid & 3;
    const float* wrow = W3 + (size_t)k * HID;
    const float* fr = h2s + b * HID;
    float d = 0.f;
    #pragma unroll 4
    for (int j = 0; j < HID; ++j) d += fr[j] * wrow[j];
    out[b * 4 + k] = d + b3[k];
  }
}

// ---------------------------------------------------------------------------
extern "C" void kernel_launch(void* const* d_in, const int* in_sizes, int n_in,
                              void* d_out, int out_size, void* d_ws, size_t ws_size,
                              hipStream_t stream) {
  const float* f0 = (const float*)d_in[0];
  const float* f1 = (const float*)d_in[1];
  const float* W1 = (const float*)d_in[2];
  const float* b1 = (const float*)d_in[3];
  const float* W2 = (const float*)d_in[4];
  const float* b2 = (const float*)d_in[5];
  const float* W3 = (const float*)d_in[6];
  const float* b3 = (const float*)d_in[7];
  float* out  = (float*)d_out;
  float* heat = (float*)d_ws;            // BRN*EN = 3888 floats of scratch

  corr_heat_kernel<<<BRN * NCOMBO, 256, 0, stream>>>(f0, f1, heat);
  mlp_kernel<<<1, 256, 0, stream>>>(heat, W1, b1, W2, b2, W3, b3, out);
}
